// AnticipatorRNN_78606491451629
// MI455X (gfx1250) — compile-verified, matrix operand order checked by reading, not run on hardware
//
#include <hip/hip_runtime.h>
#include <hip/hip_bf16.h>

typedef __attribute__((ext_vector_type(16))) _Float16 v16h;
typedef __attribute__((ext_vector_type(8)))  _Float16 v8h;
typedef __attribute__((ext_vector_type(8)))  float    v8f;

#define WMMA_F16(a, b, c) \
  __builtin_amdgcn_wmma_f32_16x16x32_f16(false, (a), false, (b), (short)0, (c), false, false)

#define JOIN16(lo, hi) \
  __builtin_shufflevector((lo), (hi), 0, 1, 2, 3, 4, 5, 6, 7, 8, 9, 10, 11, 12, 13, 14, 15)

// K index held by fragment element e for lane-half `half` (16-bit A layout, ISA 7.12.2)
__device__ __forceinline__ int kmap(int e, int half) { return e + 8 * half + (e & 8); }

// branchless fast activations (native v_exp_f32)
__device__ __forceinline__ float sig_(float x)  { return 1.0f / (1.0f + __expf(-x)); }
__device__ __forceinline__ float tanh_(float x) { return 2.0f / (1.0f + __expf(-2.0f * x)) - 1.0f; }

// ---------------------------------------------------------------- prep kernels
__global__ void zero_u32_kernel(unsigned* __restrict__ p, int n) {
  int i = blockIdx.x * blockDim.x + threadIdx.x;
  if (i < n) p[i] = 0u;
}

// frames f32 NHWC(C=3) -> f16 NHWC(C=4, ch3 = 0)
__global__ void frames_pad4_kernel(const float* __restrict__ src, _Float16* __restrict__ dst,
                                   int npix) {
  int i = blockIdx.x * blockDim.x + threadIdx.x;
  if (i < npix * 4) {
    int p = i >> 2, c = i & 3;
    dst[i] = (c < 3) ? (_Float16)src[p * 3 + c] : (_Float16)0.0f;
  }
}

// generic: pack row-major f32 W[Kreal][N] into WMMA B-fragment layout
__global__ void pack_b_frag_kernel(const float* __restrict__ src, _Float16* __restrict__ dst,
                                   int Kreal, int KT, int N, int total) {
  int idx = blockIdx.x * blockDim.x + threadIdx.x;
  if (idx >= total) return;
  int e = idx & 15, lane = (idx >> 4) & 31;
  int tile = idx >> 9;
  int kt = tile % KT, ntile = tile / KT;
  int k = kt * 32 + kmap(e, lane >> 4);
  int col = ntile * 16 + (lane & 15);
  dst[idx] = (k < Kreal) ? (_Float16)src[(size_t)k * N + col] : (_Float16)0.0f;
}

// conv1 weights [4][4][3][32] -> fragment layout for padded-Cin=4 K order (K=64)
__global__ void pack_w1_frag_kernel(const float* __restrict__ src, _Float16* __restrict__ dst) {
  int idx = blockIdx.x * blockDim.x + threadIdx.x;   // total = 2*2*512
  if (idx >= 2 * 2 * 512) return;
  int e = idx & 15, lane = (idx >> 4) & 31;
  int tile = idx >> 9;
  int kt = tile % 2, ntile = tile / 2;
  int k = kt * 32 + kmap(e, lane >> 4);
  int ci = k & 3, p = k >> 2;
  int col = ntile * 16 + (lane & 15);
  dst[idx] = (ci < 3) ? (_Float16)src[(p * 3 + ci) * 32 + col] : (_Float16)0.0f;
}

// k_l1 [1538][2048] -> fragment layout, K remapped to [enc(1024)|h1(512)|act(2)|pad(30)]
__global__ void pack_l1_frag_kernel(const float* __restrict__ src, _Float16* __restrict__ dst,
                                    int total) {
  int idx = blockIdx.x * blockDim.x + threadIdx.x;   // total = 128*49*512
  if (idx >= total) return;
  int e = idx & 15, lane = (idx >> 4) & 31;
  int tile = idx >> 9;
  int kt = tile % 49, ntile = tile / 49;
  int k = kt * 32 + kmap(e, lane >> 4);
  int col = ntile * 16 + (lane & 15);
  int srcRow;
  if (k < 1024)      srcRow = k;
  else if (k < 1536) srcRow = 1026 + (k - 1024);
  else if (k < 1538) srcRow = 1024 + (k - 1536);
  else               srcRow = -1;
  dst[idx] = (srcRow >= 0) ? (_Float16)src[(size_t)srcRow * 2048 + col] : (_Float16)0.0f;
}

// ---------------------------------------------------------------- conv (implicit GEMM, WMMA)
template <int Cin, int Cout, int IH, int IW, int OH, int OW>
__global__ void conv_wmma_kernel(const _Float16* __restrict__ X, const _Float16* __restrict__ Wp,
                                 const float* __restrict__ bias, _Float16* __restrict__ Y,
                                 int M, int tiles) {
  const int KT = Cin / 2;
  int wave = (int)((blockIdx.x * blockDim.x + threadIdx.x) >> 5);
  if (wave >= tiles) return;
  const int NT = Cout / 16;
  int ntile = wave % NT, mtile = wave / NT;
  int lane = threadIdx.x & 31;
  int nl = lane & 15, half = lane >> 4;

  int m = mtile * 16 + nl;
  int img = m / (OH * OW);
  int rem = m - img * (OH * OW);
  int oh = rem / OW, ow = rem - oh * OW;
  const _Float16* Xb = X + (((size_t)img * IH + oh * 2) * IW + ow * 2) * Cin;
  const _Float16* Wb = Wp + (size_t)ntile * KT * 512 + (size_t)lane * 16;
  const int n0 = ntile * 16;

  auto loadA = [&](int kt) -> v16h {
    int k0 = kt * 32 + 8 * half;
    int kh0 = k0 / (4 * Cin), r0 = k0 - kh0 * (4 * Cin);
    int k1 = k0 + 16;
    int kh1 = k1 / (4 * Cin), r1 = k1 - kh1 * (4 * Cin);
    v8h lo = *(const v8h*)&Xb[(size_t)kh0 * IW * Cin + r0];
    v8h hi = *(const v8h*)&Xb[(size_t)kh1 * IW * Cin + r1];
    return JOIN16(lo, hi);
  };

  v8f acc = {};
  v16h a = loadA(0);
  v16h b = *(const v16h*)&Wb[0];
  for (int kt = 0; kt < KT; ++kt) {
    v16h pa = a, pb = b;
    int kn = (kt + 1 < KT) ? kt + 1 : kt;       // software pipeline: prefetch next
    a = loadA(kn);
    b = *(const v16h*)&Wb[(size_t)kn * 512];
    acc = WMMA_F16(pa, pb, acc);
  }
#pragma unroll
  for (int r = 0; r < 8; ++r) {
    int mr = mtile * 16 + r + 8 * half;
    float y = acc[r] + bias[n0 + nl];
    y = y > 0.f ? y : 0.f;
    Y[(size_t)mr * Cout + n0 + nl] = (_Float16)y;
  }
}

// ---------------------------------------------------------------- clustered 2-layer LSTM
#define K1PAD 1568   // [enc(1024) | h1(512) | act(2) | pad(30)] = 49*32
#define KT1   49
#define H1D   512
#define K2D   768    // [h1(512) | h2(256)] = 24*32
#define KT2   24
#define H2D   256
#define BB    16
#define TT    64
#define NW    16     // workgroups in the (cluster-sized) grid

// 4-gate GEMM for one 16-column slice; explicit double-buffered B fragments
template <int KT, int GT>  // GT = n-tiles per gate
__device__ __forceinline__ void gate_gemm(const _Float16* xrow, const _Float16* wslice,
                                          int lane, int half,
                                          v8f& a0, v8f& a1, v8f& a2, v8f& a3) {
  const size_t gs = (size_t)GT * KT * 512;
  const _Float16* wb = wslice + (size_t)lane * 16;
  v16h b0 = *(const v16h*)(wb);
  v16h b1 = *(const v16h*)(wb + gs);
  v16h b2 = *(const v16h*)(wb + 2 * gs);
  v16h b3 = *(const v16h*)(wb + 3 * gs);
  for (int kt = 0; kt < KT; ++kt) {
    const _Float16* xr = xrow + kt * 32 + 8 * half;
    v16h a = JOIN16(*(const v8h*)xr, *(const v8h*)(xr + 16));
    v16h p0 = b0, p1 = b1, p2 = b2, p3 = b3;
    int kn = (kt + 1 < KT) ? kt + 1 : kt;
    const _Float16* wn = wb + (size_t)kn * 512;
    b0 = *(const v16h*)(wn);
    b1 = *(const v16h*)(wn + gs);
    b2 = *(const v16h*)(wn + 2 * gs);
    b3 = *(const v16h*)(wn + 3 * gs);
    a0 = WMMA_F16(a, p0, a0);
    a1 = WMMA_F16(a, p1, a1);
    a2 = WMMA_F16(a, p2, a2);
    a3 = WMMA_F16(a, p3, a3);
  }
}

// cluster-aware grid barrier: cluster barrier (NOP outside cluster dispatch) + atomic gate
__device__ __forceinline__ void grid_sync(unsigned* cnt, unsigned target) {
  __threadfence();
  __syncthreads();
  __builtin_amdgcn_s_cluster_barrier();     // s_barrier_signal/-wait -3 (fast path in cluster)
  if (threadIdx.x == 0) {
    __hip_atomic_fetch_add(cnt, 1u, __ATOMIC_RELEASE, __HIP_MEMORY_SCOPE_AGENT);
    while (__hip_atomic_load(cnt, __ATOMIC_ACQUIRE, __HIP_MEMORY_SCOPE_AGENT) < target)
      __builtin_amdgcn_s_sleep(1);
  }
  __syncthreads();
}

__global__ void lstm_kernel(const _Float16* __restrict__ enc,      // [16*64][1024]
                            const float*    __restrict__ actions,  // [16][64][2]
                            const _Float16* __restrict__ Wl1p,     // packed [128][49][32][16]
                            const float*    __restrict__ bl1,      // [2048]
                            const _Float16* __restrict__ Wl2p,     // packed [64][24][32][16]
                            const float*    __restrict__ bl2,      // [1024]
                            _Float16*       __restrict__ hg1,      // [16][512] shared h1
                            _Float16*       __restrict__ hg2,      // [16][256] shared h2
                            unsigned*       __restrict__ cnt,      // grid-barrier counter
                            _Float16*       __restrict__ hs)       // [16][64][256]
{
  extern __shared__ char smem[];
  _Float16* x1 = (_Float16*)(smem);                          // [16][1568]
  _Float16* x2 = (_Float16*)(smem + 50176);                  // [16][768]
  float*    c1 = (float*)   (smem + 50176 + 24576);          // [16][32] (own 2 slices)
  float*    c2 = (float*)   (smem + 50176 + 24576 + 2048);   // [16][16] (own slice)

  const int tid  = threadIdx.x;          // 0..63 (2 waves)
  const int lane = tid & 31;
  const int wid  = tid >> 5;
  const int nl   = lane & 15, half = lane >> 4;
  const int wg   = blockIdx.x;           // 0..15, owns layer1 slices 2wg,2wg+1; layer2 slice wg
  unsigned ep = 0;

  for (int i = tid; i < BB * 32; i += 64) c1[i] = 0.f;
  for (int i = tid; i < BB * 16; i += 64) c2[i] = 0.f;
  // zero x1 pad columns (1538..1567)
  for (int i = tid; i < BB * 32; i += 64) {
    int m = i >> 5, c = i & 31;
    if (c >= 2) x1[m * K1PAD + 1536 + c] = (_Float16)0.0f;
  }
  __syncthreads();

  for (int t = 0; t < TT; ++t) {
    // stage x1 = [enc(t) | h1_prev | act(t)], x2 = [h1_prev | h2_prev]
    for (int i = tid; i < BB * 128; i += 64) {
      int m = i >> 7, c8 = i & 127;
      *(v8h*)&x1[m * K1PAD + c8 * 8] =
          *(const v8h*)&enc[((size_t)(m * TT + t) << 10) + c8 * 8];
    }
    for (int i = tid; i < BB * 64; i += 64) {
      int m = i >> 6, c8 = i & 63;
      v8h v = *(const v8h*)&hg1[m * H1D + c8 * 8];
      *(v8h*)&x1[m * K1PAD + 1024 + c8 * 8] = v;
      *(v8h*)&x2[m * K2D + c8 * 8] = v;
    }
    for (int i = tid; i < BB * 32; i += 64) {
      int m = i >> 5, c8 = i & 31;
      *(v8h*)&x2[m * K2D + 512 + c8 * 8] = *(const v8h*)&hg2[m * H2D + c8 * 8];
    }
    if (tid < BB * 2) {
      int m = tid >> 1, a = tid & 1;
      x1[m * K1PAD + 1536 + a] = (_Float16)actions[(m * TT + t) * 2 + a];
    }
    __syncthreads();

    // ---- layer 1: this wave owns slice (2*wg + wid)
    {
      int s = wg * 2 + wid;
      int n0 = s * 16;
      v8f g0 = {}, g1 = {}, g2 = {}, g3 = {};
      gate_gemm<KT1, 32>(&x1[nl * K1PAD], Wl1p + (size_t)s * KT1 * 512, lane, half,
                         g0, g1, g2, g3);
#pragma unroll
      for (int r = 0; r < 8; ++r) {
        int m = r + 8 * half;
        int col = n0 + nl;
        float ig = g0[r] + bl1[0 * H1D + col];
        float jg = g1[r] + bl1[1 * H1D + col];
        float fg = g2[r] + bl1[2 * H1D + col];
        float og = g3[r] + bl1[3 * H1D + col];
        int cc = m * 32 + wid * 16 + nl;
        float cn = sig_(fg + 1.0f) * c1[cc] + sig_(ig) * tanh_(jg);
        c1[cc] = cn;
        hg1[m * H1D + col] = (_Float16)(sig_(og) * tanh_(cn));
      }
    }
    grid_sync(cnt, ++ep * NW);

    // restage full h1_new into x2
    for (int i = tid; i < BB * 64; i += 64) {
      int m = i >> 6, c8 = i & 63;
      *(v8h*)&x2[m * K2D + c8 * 8] = *(const v8h*)&hg1[m * H1D + c8 * 8];
    }
    __syncthreads();

    // ---- layer 2: wave 0 owns slice wg
    if (wid == 0) {
      int n0 = wg * 16;
      v8f g0 = {}, g1 = {}, g2 = {}, g3 = {};
      gate_gemm<KT2, 16>(&x2[nl * K2D], Wl2p + (size_t)wg * KT2 * 512, lane, half,
                         g0, g1, g2, g3);
#pragma unroll
      for (int r = 0; r < 8; ++r) {
        int m = r + 8 * half;
        int col = n0 + nl;
        float ig = g0[r] + bl2[0 * H2D + col];
        float jg = g1[r] + bl2[1 * H2D + col];
        float fg = g2[r] + bl2[2 * H2D + col];
        float og = g3[r] + bl2[3 * H2D + col];
        int cc = m * 16 + nl;
        float cn = sig_(fg + 1.0f) * c2[cc] + sig_(ig) * tanh_(jg);
        c2[cc] = cn;
        _Float16 hv = (_Float16)(sig_(og) * tanh_(cn));
        hg2[m * H2D + col] = hv;
        hs[((size_t)(m * TT + t) << 8) + col] = hv;
      }
    }
    grid_sync(cnt, ++ep * NW);
  }
}

// ---------------------------------------------------------------- dense head
__global__ void head1_kernel(const _Float16* __restrict__ hsm,  // [1024][256]
                             const _Float16* __restrict__ Wd1p, // packed [8][8][32][16]
                             const float* __restrict__ bd1,
                             float* __restrict__ d1, int tiles) {
  int wave = (int)((blockIdx.x * blockDim.x + threadIdx.x) >> 5);
  if (wave >= tiles) return;
  const int NT = 128 / 16;
  int ntile = wave % NT, mtile = wave / NT;
  int lane = threadIdx.x & 31, nl = lane & 15, half = lane >> 4;
  int n0 = ntile * 16;
  const _Float16* Ab = &hsm[(size_t)(mtile * 16 + nl) * 256 + 8 * half];
  const _Float16* Wb = &Wd1p[(size_t)ntile * 8 * 512 + (size_t)lane * 16];
  v8f acc = {};
  v16h a = JOIN16(*(const v8h*)Ab, *(const v8h*)(Ab + 16));
  v16h b = *(const v16h*)Wb;
  for (int kt = 0; kt < 8; ++kt) {
    v16h pa = a, pb = b;
    int kn = (kt + 1 < 8) ? kt + 1 : kt;
    const _Float16* an = Ab + kn * 32;
    a = JOIN16(*(const v8h*)an, *(const v8h*)(an + 16));
    b = *(const v16h*)&Wb[(size_t)kn * 512];
    acc = WMMA_F16(pa, pb, acc);
  }
#pragma unroll
  for (int r = 0; r < 8; ++r) {
    float y = acc[r] + bd1[n0 + nl];
    d1[(size_t)(mtile * 16 + r + 8 * half) * 128 + n0 + nl] = y > 0.f ? y : 0.f;
  }
}

__global__ void head2_kernel(const float* __restrict__ d1, const float* __restrict__ wd2,
                             const float* __restrict__ bd2, float* __restrict__ out, int M) {
  int wave = (int)((blockIdx.x * blockDim.x + threadIdx.x) >> 5);
  int lane = threadIdx.x & 31;
  if (wave >= M) return;
  float s = 0.f;
#pragma unroll
  for (int q = 0; q < 4; ++q) {
    int j = lane + 32 * q;
    s += d1[(size_t)wave * 128 + j] * wd2[j];
  }
  for (int off = 16; off > 0; off >>= 1) s += __shfl_xor(s, off, 32);
  if (lane == 0) out[wave] = s + bd2[0];
}

// ---------------------------------------------------------------- launch
extern "C" void kernel_launch(void* const* d_in, const int* in_sizes, int n_in,
                              void* d_out, int out_size, void* d_ws, size_t ws_size,
                              hipStream_t stream) {
  (void)in_sizes; (void)n_in; (void)out_size; (void)ws_size;
  const float* frames  = (const float*)d_in[0];
  const float* actions = (const float*)d_in[1];
  const float* w1 = (const float*)d_in[2];  const float* b1 = (const float*)d_in[3];
  const float* w2 = (const float*)d_in[4];  const float* b2 = (const float*)d_in[5];
  const float* w3 = (const float*)d_in[6];  const float* b3 = (const float*)d_in[7];
  const float* w4 = (const float*)d_in[8];  const float* b4 = (const float*)d_in[9];
  const float* kl1 = (const float*)d_in[10]; const float* bl1 = (const float*)d_in[11];
  const float* kl2 = (const float*)d_in[12]; const float* bl2 = (const float*)d_in[13];
  const float* wd1 = (const float*)d_in[14]; const float* bd1 = (const float*)d_in[15];
  const float* wd2 = (const float*)d_in[16]; const float* bd2 = (const float*)d_in[17];
  float* out = (float*)d_out;

  char* ws = (char*)d_ws;
  size_t off = 0;
  auto alloc = [&](size_t bytes) -> char* {
    char* p = ws + off; off += (bytes + 255) & ~(size_t)255; return p;
  };
  _Float16* x0   = (_Float16*)alloc((size_t)1024 * 64 * 64 * 4 * 2);
  _Float16* y1   = (_Float16*)alloc((size_t)1024 * 31 * 31 * 32 * 2);
  _Float16* y2   = (_Float16*)alloc((size_t)1024 * 14 * 14 * 64 * 2);
  _Float16* y3   = (_Float16*)alloc((size_t)1024 * 6 * 6 * 128 * 2);
  _Float16* y4   = (_Float16*)alloc((size_t)1024 * 1024 * 2);          // enc
  _Float16* wp1  = (_Float16*)alloc((size_t)2 * 2 * 512 * 2);
  _Float16* wp2  = (_Float16*)alloc((size_t)4 * 16 * 512 * 2);
  _Float16* wp3  = (_Float16*)alloc((size_t)8 * 32 * 512 * 2);
  _Float16* wp4  = (_Float16*)alloc((size_t)16 * 64 * 512 * 2);
  _Float16* wl1p = (_Float16*)alloc((size_t)128 * 49 * 512 * 2);
  _Float16* wl2p = (_Float16*)alloc((size_t)64 * 24 * 512 * 2);
  _Float16* wd1p = (_Float16*)alloc((size_t)8 * 8 * 512 * 2);
  _Float16* hsb  = (_Float16*)alloc((size_t)1024 * 256 * 2);
  float*    d1b  = (float*)   alloc((size_t)1024 * 128 * 4);
  char*     lst  = alloc(16384 + 8192 + 256);                          // hg1 | hg2 | cnt
  _Float16* hg1  = (_Float16*)lst;
  _Float16* hg2  = (_Float16*)(lst + 16384);
  unsigned* cnt  = (unsigned*)(lst + 16384 + 8192);

  // ---- prep
  zero_u32_kernel<<<(6145 + 255) / 256, 256, 0, stream>>>((unsigned*)lst, 6145);
  {
    int n = 1024 * 64 * 64 * 4;
    frames_pad4_kernel<<<(n + 255) / 256, 256, 0, stream>>>(frames, x0, 1024 * 64 * 64);
  }
  pack_w1_frag_kernel<<<(2048 + 255) / 256, 256, 0, stream>>>(w1, wp1);
  { int tot = 4 * 16 * 512;
    pack_b_frag_kernel<<<(tot + 255) / 256, 256, 0, stream>>>(w2, wp2, 512, 16, 64, tot); }
  { int tot = 8 * 32 * 512;
    pack_b_frag_kernel<<<(tot + 255) / 256, 256, 0, stream>>>(w3, wp3, 1024, 32, 128, tot); }
  { int tot = 16 * 64 * 512;
    pack_b_frag_kernel<<<(tot + 255) / 256, 256, 0, stream>>>(w4, wp4, 2048, 64, 256, tot); }
  { int tot = 128 * 49 * 512;
    pack_l1_frag_kernel<<<(tot + 255) / 256, 256, 0, stream>>>(kl1, wl1p, tot); }
  { int tot = 64 * 24 * 512;
    pack_b_frag_kernel<<<(tot + 255) / 256, 256, 0, stream>>>(kl2, wl2p, 768, 24, 1024, tot); }
  { int tot = 8 * 8 * 512;
    pack_b_frag_kernel<<<(tot + 255) / 256, 256, 0, stream>>>(wd1, wd1p, 256, 8, 128, tot); }

  // ---- conv encoder
  { int M = 1024 * 31 * 31, tiles = (M / 16) * 2;
    conv_wmma_kernel<4, 32, 64, 64, 31, 31>
        <<<(tiles * 32 + 255) / 256, 256, 0, stream>>>(x0, wp1, b1, y1, M, tiles); }
  { int M = 1024 * 14 * 14, tiles = (M / 16) * 4;
    conv_wmma_kernel<32, 64, 31, 31, 14, 14>
        <<<(tiles * 32 + 255) / 256, 256, 0, stream>>>(y1, wp2, b2, y2, M, tiles); }
  { int M = 1024 * 6 * 6, tiles = (M / 16) * 8;
    conv_wmma_kernel<64, 128, 14, 14, 6, 6>
        <<<(tiles * 32 + 255) / 256, 256, 0, stream>>>(y2, wp3, b3, y3, M, tiles); }
  { int M = 1024 * 2 * 2, tiles = (M / 16) * 16;
    conv_wmma_kernel<128, 256, 6, 6, 2, 2>
        <<<(tiles * 32 + 255) / 256, 256, 0, stream>>>(y3, wp4, b4, y4, M, tiles); }

  // ---- clustered LSTM: 16 WGs (one cluster's worth), 2 waves each, LDS-resident c-state
  {
    size_t smem = 50176 + 24576 + 2048 + 1024;   // x1 | x2 | c1 | c2 = 77,824 B
    lstm_kernel<<<NW, 64, smem, stream>>>(y4, actions, wl1p, bl1, wl2p, bl2,
                                          hg1, hg2, cnt, hsb);
  }

  // ---- dense head
  { int tiles = (1024 / 16) * (128 / 16);
    head1_kernel<<<(tiles * 32 + 255) / 256, 256, 0, stream>>>(hsb, wd1p, bd1, d1b, tiles); }
  head2_kernel<<<(1024 * 32 + 255) / 256, 256, 0, stream>>>(d1b, wd2, bd2, out, 1024);
}
